// MyAttentionEncoder_5720896438770
// MI455X (gfx1250) — matrix operand order repaired and checked by reading, not compiled, not run on hardware
//
#include <hip/hip_runtime.h>
#include <hip/hip_bf16.h>
#include <math.h>

typedef __attribute__((ext_vector_type(16))) _Float16 v16h;
typedef __attribute__((ext_vector_type(8)))  float    v8f;

#define B_    32
#define L_    512
#define NL_   3
#define H_    8
#define DK_   25
#define DV_   25
#define DM_   200
#define DFF_  800
#define WD_   150
#define TD_   50
#define EPS_  1e-3f
#define BL_   (B_ * L_)
#define DMP_  224     // d_model padded to 7*32 (A-operand stride, zero pad)
#define QKP_  256     // H_ * 32 : padded per-head strip width for q/k/v

// ---------------------------------------------------------------------------
// Embedding + bias + positional encoding (writes padded-stride x; pad cols
// pre-zeroed by memset).
// ---------------------------------------------------------------------------
__global__ __launch_bounds__(256) void embed_kernel(
    const float* __restrict__ we, const float* __restrict__ ee,
    const float* __restrict__ te, const float* __restrict__ pt,
    const float* __restrict__ bias,
    const int* __restrict__ wseq, const int* __restrict__ ews,
    const int* __restrict__ tags, const int* __restrict__ pos,
    float* __restrict__ x)
{
    int tok = blockIdx.x;
    int c   = threadIdx.x;
    if (c >= DM_) return;
    int w = wseq[tok], e = ews[tok], t = tags[tok], p = pos[tok];
    float v;
    if (c < WD_) v = we[(size_t)w * WD_ + c] + ee[(size_t)e * WD_ + c];
    else         v = te[(size_t)t * TD_ + (c - WD_)];
    v += bias[c] + pt[(size_t)p * DM_ + c];
    x[(size_t)tok * DMP_ + c] = v;
}

// ---------------------------------------------------------------------------
// Weight packers -> f16 WMMA B-operand register layout, 4-tile interleaved:
//   Bp[tilegroup][kchunk][j=tile%4][lane][16 halves]
//   half i -> K = kc*32 + 16*(lane>>4) + i ;  N = nt*16 + (lane&15)
// Clamped (unconditional) loads + select; tiles beyond Nreal pack as zeros so
// the GEMM needs no B-side guards.
// ---------------------------------------------------------------------------
__global__ __launch_bounds__(32) void pack_wT(          // W stored [Nreal,Kreal]
    const float* __restrict__ W, v16h* __restrict__ Bp,
    int Nreal, int Kreal, int nChunks)
{
    int lane = threadIdx.x, g = lane >> 4, lm = lane & 15;
    int nt = blockIdx.x, kc = blockIdx.y;
    int n  = nt * 16 + lm;
    int ncl = n < Nreal ? n : Nreal - 1;
    v16h b;
#pragma unroll
    for (int i = 0; i < 16; ++i) {
        int k   = kc * 32 + 16 * g + i;
        int kcl = k < Kreal ? k : Kreal - 1;
        float f = W[(size_t)ncl * Kreal + kcl];
        if (n >= Nreal || k >= Kreal) f = 0.0f;
        b[i] = (_Float16)f;
    }
    Bp[(((size_t)(nt >> 2) * nChunks + kc) * 4 + (nt & 3)) * 32 + lane] = b;
}

__global__ __launch_bounds__(32) void pack_qkv(         // W [H][DM][DK], N=h*32+dv
    const float* __restrict__ W, v16h* __restrict__ Bp, int nChunks)
{
    int lane = threadIdx.x, g = lane >> 4, lm = lane & 15;
    int nt = blockIdx.x, kc = blockIdx.y;
    int n  = nt * 16 + lm;
    int hd = n >> 5, dv = n & 31;
    int dvc = dv < DK_ ? dv : DK_ - 1;
    const float* Wh = W + (size_t)hd * DM_ * DK_;
    v16h b;
#pragma unroll
    for (int i = 0; i < 16; ++i) {
        int k   = kc * 32 + 16 * g + i;
        int kcl = k < DM_ ? k : DM_ - 1;
        float f = Wh[(size_t)kcl * DK_ + dvc];
        if (dv >= DK_ || k >= DM_) f = 0.0f;
        b[i] = (_Float16)f;
    }
    Bp[(((size_t)(nt >> 2) * nChunks + kc) * 4 + (nt & 3)) * 32 + lane] = b;
}

// ---------------------------------------------------------------------------
// GEMM, 4-way N-blocked: one wave computes a 16x64 C strip (4 WMMA tiles).
// A (fp32, padded lda) loaded & converted once per K-chunk; the 4 B operands
// for each chunk are contiguous (one base pointer, immediate offsets), so the
// 8 b128 B-loads issue as one clause overlapping the 4 WMMAs.
//   C = act( scale * (A x B) + bias ) [+ R]
// Packed B covers 4*gridDim.x tiles (zero-padded).  Kpad % 32 == 0.
// ---------------------------------------------------------------------------
__global__ __launch_bounds__(32) void gemm16p4(
    const float* __restrict__ A, const v16h* __restrict__ Bp,
    const float* __restrict__ bias, const float* __restrict__ R,
    float* __restrict__ C,
    int N, int nChunks, int lda, int ldc, int ldr,
    int relu, float scale)
{
    int lane = threadIdx.x, g = lane >> 4, lm = lane & 15;
    int nt0 = blockIdx.x * 4;          // first of 4 column tiles
    int tm  = blockIdx.y * 16;

    const float* ar = A + (size_t)(tm + lm) * lda + 16 * g;
    const v16h*  bp = Bp + (size_t)blockIdx.x * nChunks * 4 * 32 + lane;

    v8f acc[4] = {{}, {}, {}, {}};
    for (int c = 0; c < nChunks; ++c) {
        const float* ap = ar + (c << 5);
        const v16h*  bc = bp + (size_t)c * 4 * 32;
        v16h b0 = bc[0];               // byte offsets 0 / 1024 / 2048 / 3072
        v16h b1 = bc[32];
        v16h b2 = bc[64];
        v16h b3 = bc[96];
        v16h a;
#pragma unroll
        for (int i = 0; i < 8; ++i) a[i] = (_Float16)ap[i];          // K=k0+16g+i
#pragma unroll
        for (int i = 0; i < 8; ++i) a[i + 8] = (_Float16)ap[8 + i]; // K=k0+16g+8+i
        acc[0] = __builtin_amdgcn_wmma_f32_16x16x32_f16(
                     false, a, false, b0, (short)0, acc[0], false, false);
        acc[1] = __builtin_amdgcn_wmma_f32_16x16x32_f16(
                     false, a, false, b1, (short)0, acc[1], false, false);
        acc[2] = __builtin_amdgcn_wmma_f32_16x16x32_f16(
                     false, a, false, b2, (short)0, acc[2], false, false);
        acc[3] = __builtin_amdgcn_wmma_f32_16x16x32_f16(
                     false, a, false, b3, (short)0, acc[3], false, false);
    }

    int m0 = tm + 8 * g;
#pragma unroll
    for (int j = 0; j < 4; ++j) {
        int col = (nt0 + j) * 16 + lm;
        if (col < N) {
            float bv = bias ? bias[col] : 0.0f;
#pragma unroll
            for (int r = 0; r < 8; ++r) {
                int m = m0 + r;
                float v = acc[j][r] * scale + bv;
                if (relu) v = fmaxf(v, 0.0f);
                if (R) v += R[(size_t)m * ldr + col];
                C[(size_t)m * ldc + col] = v;
            }
        }
    }
}

// ---------------------------------------------------------------------------
// Fused attention: one wave per (head, batch, 16-row Q tile).
// q/k/v: [B*L, 256] fp32, head strip at col h*32, cols 25..31 are zeros;
// 1/sqrt(d_model) folded into q.  Scores strip in LDS, probs re-stored as f16
// in A-operand-friendly runs.  Output head-major into o[B*L, DMP_].
// ---------------------------------------------------------------------------
__global__ __launch_bounds__(32) void attn_kernel(
    const float* __restrict__ q, const float* __restrict__ k_,
    const float* __restrict__ v_, const int* __restrict__ wseq,
    float* __restrict__ o)
{
    __shared__ float    sS[16][L_];   // 32 KB scores
    __shared__ _Float16 sP[16][L_];   // 16 KB probabilities (f16)
    __shared__ float    smask[L_];
    __shared__ float    sinv[16];

    int lane = threadIdx.x, g = lane >> 4, lm = lane & 15;
    int qt = blockIdx.x, b = blockIdx.y, h = blockIdx.z;
    int hc = h * 32;
    size_t rb = (size_t)b * L_;

    for (int c = lane; c < L_; c += 32) {
        int w = wseq[b * L_ + c];
        smask[c] = (w == 0) ? -1e30f : 0.0f;
    }

    // ---- Q tile A-operand (unguarded contiguous runs; pad cols are zero)
    const float* qp = q + (rb + qt * 16 + lm) * QKP_ + hc + 16 * g;
    v16h aq;
#pragma unroll
    for (int i = 0; i < 8; ++i) aq[i] = (_Float16)qp[i];
#pragma unroll
    for (int i = 0; i < 8; ++i) aq[i + 8] = (_Float16)qp[8 + i];

    // ---- scores: 32 key tiles of 16 -> LDS strip
    for (int j = 0; j < 32; ++j) {
        int key = j * 16 + lm;
        const float* kp = k_ + (rb + key) * QKP_ + hc + 16 * g;
        v16h bk;
#pragma unroll
        for (int i = 0; i < 16; ++i) bk[i] = (_Float16)kp[i];
        v8f s = {};
        s = __builtin_amdgcn_wmma_f32_16x16x32_f16(
                false, aq, false, bk, (short)0, s, false, false);
#pragma unroll
        for (int r = 0; r < 8; ++r)
            sS[r + 8 * g][key] = s[r] + smask[key];
    }

    // ---- softmax per row (lanes 0..15 own one row each)
    if (lane < 16) {
        float mx = -INFINITY;
        for (int c = 0; c < L_; ++c) mx = fmaxf(mx, sS[lane][c]);
        float sum = 0.0f;
        for (int c = 0; c < L_; ++c) {
            float p = __expf(sS[lane][c] - mx);
            sP[lane][c] = (_Float16)p;
            sum += p;
        }
        sinv[lane] = 1.0f / sum;
    }

    // ---- O = P x V : 16 chunks of 32 keys
    v8f o0 = {}, o1 = {};
    for (int ch = 0; ch < 16; ++ch) {
        const _Float16* pp = &sP[lm][ch * 32 + 16 * g];
        v16h ap;
#pragma unroll
        for (int i = 0; i < 8; ++i) ap[i] = pp[i];
#pragma unroll
        for (int i = 0; i < 8; ++i) ap[i + 8] = pp[8 + i];
        v16h b0, b1;
#pragma unroll
        for (int i = 0; i < 16; ++i) {
            const float* vp = v_ + (rb + ch * 32 + 16 * g + i) * QKP_ + hc;
            b0[i] = (_Float16)vp[lm];
            b1[i] = (_Float16)vp[16 + lm];
        }
        o0 = __builtin_amdgcn_wmma_f32_16x16x32_f16(
                 false, ap, false, b0, (short)0, o0, false, false);
        o1 = __builtin_amdgcn_wmma_f32_16x16x32_f16(
                 false, ap, false, b1, (short)0, o1, false, false);
    }

    // ---- scale by 1/rowsum, store head-major concat [B*L, DMP_]
#pragma unroll
    for (int r = 0; r < 8; ++r) {
        int ml = r + 8 * g;
        size_t orow = (rb + qt * 16 + ml) * DMP_ + (size_t)h * DV_;
        float iv = sinv[ml];
        o[orow + lm] = o0[r] * iv;
        if (16 + lm < DV_) o[orow + 16 + lm] = o1[r] * iv;
    }
}

// ---------------------------------------------------------------------------
// Custom LayerNorm (ddof=1 variance, eps added to sigma). One wave per token.
// ---------------------------------------------------------------------------
__global__ __launch_bounds__(32) void layernorm_kernel(
    const float* __restrict__ X, int ldx,
    const float* __restrict__ ga, const float* __restrict__ gb,
    float* __restrict__ Y, int ldy, int D)
{
    int tok  = blockIdx.x;
    int lane = threadIdx.x;
    const float* xr = X + (size_t)tok * ldx;

    float s = 0.0f;
    for (int c = lane; c < D; c += 32) s += xr[c];
#pragma unroll
    for (int off = 16; off; off >>= 1) s += __shfl_xor(s, off, 32);
    float mu = s / (float)D;

    float vs = 0.0f;
    for (int c = lane; c < D; c += 32) { float d = xr[c] - mu; vs += d * d; }
#pragma unroll
    for (int off = 16; off; off >>= 1) vs += __shfl_xor(vs, off, 32);
    float inv = 1.0f / (sqrtf(vs / (float)(D - 1)) + EPS_);

    float* yr = Y + (size_t)tok * ldy;
    for (int c = lane; c < D; c += 32)
        yr[c] = (xr[c] - mu) * inv * ga[c] + gb[c];
}

// ---------------------------------------------------------------------------
extern "C" void kernel_launch(void* const* d_in, const int* in_sizes, int n_in,
                              void* d_out, int out_size, void* d_ws, size_t ws_size,
                              hipStream_t stream)
{
    const float* word_emb    = (const float*)d_in[0];
    const float* extword_emb = (const float*)d_in[1];
    const float* tag_emb     = (const float*)d_in[2];
    const float* pos_table   = (const float*)d_in[3];
    const float* bias        = (const float*)d_in[4];
    const float* w_qs        = (const float*)d_in[5];
    const float* w_ks        = (const float*)d_in[6];
    const float* w_vs        = (const float*)d_in[7];
    const float* w_o         = (const float*)d_in[8];
    const float* ffn_w1      = (const float*)d_in[9];
    const float* ffn_b1      = (const float*)d_in[10];
    const float* ffn_w2      = (const float*)d_in[11];
    const float* ffn_b2      = (const float*)d_in[12];
    const float* ln_ffn_a    = (const float*)d_in[13];
    const float* ln_ffn_b    = (const float*)d_in[14];
    const float* ln_attn_a   = (const float*)d_in[15];
    const float* ln_attn_b   = (const float*)d_in[16];
    const int*   word_seq    = (const int*)d_in[17];
    const int*   extwords    = (const int*)d_in[18];
    const int*   tags        = (const int*)d_in[19];
    const int*   pos_seq     = (const int*)d_in[20];
    float* out = (float*)d_out;

    // ---- workspace carve-up: fp32 region, then 32B-aligned f16 pack region
    // packed tile counts padded to multiples of 4 (N-blocking)
    float* x  = (float*)d_ws;                         // [BL, DMP]
    float* t  = x  + (size_t)BL_ * DMP_;              // [BL, DM]
    float* hb = t  + (size_t)BL_ * DM_;               // [BL, DFF]
    float* qb = hb + (size_t)BL_ * DFF_;              // [BL, QKP]
    float* kb = qb + (size_t)BL_ * QKP_;              // [BL, QKP]
    float* vb = kb + (size_t)BL_ * QKP_;              // [BL, QKP]
    float* ob = vb + (size_t)BL_ * QKP_;              // [BL, DMP]
    v16h* W1p = (v16h*)(ob + (size_t)BL_ * DMP_);     // 52 x  7 x 32
    v16h* W2p = W1p + (size_t)52 * 7  * 32;           // 16 x 25 x 32
    v16h* Wqp = W2p + (size_t)16 * 25 * 32;           // 16 x  7 x 32
    v16h* Wkp = Wqp + (size_t)16 * 7  * 32;
    v16h* Wvp = Wkp + (size_t)16 * 7  * 32;
    v16h* Wop = Wvp + (size_t)16 * 7  * 32;           // 16 x  7 x 32

    const float inv_temper = 1.0f / sqrtf((float)DM_);
    const int mt = BL_ / 16;                          // 1024 row tiles

    // zero padded activation buffers once (pad columns stay zero thereafter)
    hipMemsetAsync(x,  0, (size_t)BL_ * DMP_ * sizeof(float), stream);
    hipMemsetAsync(ob, 0, (size_t)BL_ * DMP_ * sizeof(float), stream);

    embed_kernel<<<BL_, 256, 0, stream>>>(
        word_emb, extword_emb, tag_emb, pos_table, bias,
        word_seq, extwords, tags, pos_seq, x);

    for (int i = 0; i < NL_; ++i) {
        // ---- pack this layer's weights into WMMA B-operand layout
        pack_wT <<<dim3(52, 7),  32, 0, stream>>>(ffn_w1 + (size_t)i*DFF_*DM_, W1p, DFF_, DM_, 7);
        pack_wT <<<dim3(16, 25), 32, 0, stream>>>(ffn_w2 + (size_t)i*DM_*DFF_, W2p, DM_, DFF_, 25);
        pack_qkv<<<dim3(16, 7),  32, 0, stream>>>(w_qs  + (size_t)i*H_*DM_*DK_, Wqp, 7);
        pack_qkv<<<dim3(16, 7),  32, 0, stream>>>(w_ks  + (size_t)i*H_*DM_*DK_, Wkp, 7);
        pack_qkv<<<dim3(16, 7),  32, 0, stream>>>(w_vs  + (size_t)i*H_*DM_*DK_, Wvp, 7);
        pack_wT <<<dim3(16, 7),  32, 0, stream>>>(w_o   + (size_t)i*DM_*(H_*DV_), Wop, DM_, H_*DV_, 7);

        // ---- FFN first: h = relu(x W1^T + b1)
        gemm16p4<<<dim3(13, mt), 32, 0, stream>>>(
            x, W1p, ffn_b1 + (size_t)i*DFF_, nullptr, hb,
            DFF_, 7, DMP_, DFF_, 0, 1, 1.0f);
        // ---- t = h W2^T + b2 + x ; x = LN_ffn(t)
        gemm16p4<<<dim3(4, mt), 32, 0, stream>>>(
            hb, W2p, ffn_b2 + (size_t)i*DM_, x, t,
            DM_, 25, DFF_, DM_, DMP_, 0, 1.0f);
        layernorm_kernel<<<BL_, 32, 0, stream>>>(
            t, DM_, ln_ffn_a + (size_t)i*DM_, ln_ffn_b + (size_t)i*DM_, x, DMP_, DM_);

        // ---- fused Q/K/V projections (all heads, N=256, pad cols -> zeros)
        gemm16p4<<<dim3(4, mt), 32, 0, stream>>>(
            x, Wqp, nullptr, nullptr, qb, QKP_, 7, DMP_, QKP_, 0, 0, inv_temper);
        gemm16p4<<<dim3(4, mt), 32, 0, stream>>>(
            x, Wkp, nullptr, nullptr, kb, QKP_, 7, DMP_, QKP_, 0, 0, 1.0f);
        gemm16p4<<<dim3(4, mt), 32, 0, stream>>>(
            x, Wvp, nullptr, nullptr, vb, QKP_, 7, DMP_, QKP_, 0, 0, 1.0f);

        // ---- fused softmax attention
        attn_kernel<<<dim3(L_/16, B_, H_), 32, 0, stream>>>(qb, kb, vb, word_seq, ob);

        // ---- t = o W_o^T + x ; x (or out) = LN_attn(t)
        gemm16p4<<<dim3(4, mt), 32, 0, stream>>>(
            ob, Wop, nullptr, x, t, DM_, 7, DMP_, DM_, DMP_, 0, 1.0f);
        float* dst = (i == NL_ - 1) ? out : x;
        int ldd    = (i == NL_ - 1) ? DM_ : DMP_;
        layernorm_kernel<<<BL_, 32, 0, stream>>>(
            t, DM_, ln_attn_a + (size_t)i*DM_, ln_attn_b + (size_t)i*DM_, dst, ldd, DM_);
    }
}
